// Gtu_24086176596328
// MI455X (gfx1250) — matrix operand order repaired and checked by reading, not compiled, not run on hardware
//
#include <hip/hip_runtime.h>
#include <hip/hip_bf16.h>

// ---------------------------------------------------------------------------
// GTU (gated Toeplitz unit) for MI455X / gfx1250, all heavy math on
// v_wmma_f32_16x16x32_f16 (wave32 WMMA, fp32 accumulate).
//
// FFT circular conv == Toeplitz matmul:  out[i] = sum_j T[(i-j)&127] v[j]
// so each direction is 768 batched GEMMs (M=64,K=64,N=512) -> WMMA.
// K is a template parameter so every GEMM body fully unrolls into
// straight-line code: no loop-carried register rotation, and the scheduler
// hoists next-step fragment loads above current-step WMMAs freely.
// ---------------------------------------------------------------------------

typedef __attribute__((ext_vector_type(16))) _Float16 v16h;
typedef __attribute__((ext_vector_type(8)))  _Float16 v8h;
typedef __attribute__((ext_vector_type(8)))  float    v8f;

#define GTU_EPS 1e-6f

__device__ __forceinline__ float silu_f(float x) { return x / (1.0f + __expf(-x)); }
__device__ __forceinline__ float relu_f(float x) { return x > 0.0f ? x : 0.0f; }

__device__ __forceinline__ v8f vzero8() {
  v8f z;
  #pragma unroll
  for (int i = 0; i < 8; ++i) z[i] = 0.0f;
  return z;
}

union V16HU { v16h v; v8h h[2]; };

// A fragment: 16x32 f16, row-major source, lda in halves.
// ISA 7.12.2: lane<16 -> M=lane, K={0..7,16..23}; lane>=16 -> M=lane-16, K={8..15,24..31}
__device__ __forceinline__ v16h load_a_frag(const _Float16* A, int lda, int lane) {
  int m  = lane & 15;
  int kb = (lane >> 4) << 3;   // 0 or 8
  const _Float16* p = A + m * lda + kb;
  V16HU r;
  r.h[0] = *(const v8h*)(p);        // K = kb .. kb+7
  r.h[1] = *(const v8h*)(p + 16);   // K = 16+kb .. 16+kb+7
  return r.v;
}

// B fragment: 32x16 f16 from transposed [N][K] buffer, ldb in halves.
// lanes 0-15: N=lane, K={0..7,16..23} ; lanes 16-31: N=lane-16, K={8..15,24..31}
__device__ __forceinline__ v16h load_b_frag(const _Float16* Bt, int ldb, int lane) {
  int n  = lane & 15;
  int kb = (lane >> 4) << 3;   // 0 or 8
  const _Float16* p = Bt + n * ldb + kb;
  V16HU r;
  r.h[0] = *(const v8h*)(p);
  r.h[1] = *(const v8h*)(p + 16);
  return r.v;
}

// One wave computes a 16(M) x 64(N) strip; A at tile row origin, Bt at col
// origin. Fully unrolled over K (compile-time).
template<int K>
__device__ __forceinline__ void wave_gemm16x64(const _Float16* A, int lda,
                                               const _Float16* Bt, int ldb,
                                               int lane, v8f acc[4]) {
  #pragma unroll
  for (int k0 = 0; k0 < K; k0 += 32) {
    if (k0 + 32 < K)
      __builtin_prefetch(A + k0 + 32, 0, 3);         // global_prefetch_b8 (near)
    v16h a = load_a_frag(A + k0, lda, lane);
    #pragma unroll
    for (int t = 0; t < 4; ++t) {
      v16h b = load_b_frag(Bt + (size_t)(t * 16) * ldb + k0, ldb, lane);
      acc[t] = __builtin_amdgcn_wmma_f32_16x16x32_f16(
          false, a, false, b, (short)0, acc[t], false, false);
    }
  }
}

// ---------------- pre-norm: xn = srms(x) -> f16, wave per token -------------
__global__ __launch_bounds__(256) void k_srms(const float* __restrict__ x,
                                              _Float16* __restrict__ xn) {
  int wv   = (blockIdx.x * blockDim.x + threadIdx.x) >> 5;
  int lane = threadIdx.x & 31;
  const float* row = x + (size_t)wv * 384;
  float v[12], s = 0.0f;
  #pragma unroll
  for (int j = 0; j < 12; ++j) { v[j] = row[lane + 32 * j]; s += v[j] * v[j]; }
  #pragma unroll
  for (int off = 16; off > 0; off >>= 1) s += __shfl_xor(s, off, 32);
  float r = rsqrtf(s * (1.0f / 384.0f) + GTU_EPS);
  _Float16* o = xn + (size_t)wv * 384;
  #pragma unroll
  for (int j = 0; j < 12; ++j) o[lane + 32 * j] = (_Float16)(v[j] * r);
}

// ------------- weight transpose+convert: Wt[n][k] = f16(W[k][n]) ------------
__global__ __launch_bounds__(256) void k_wt(const float* __restrict__ W,
                                            _Float16* __restrict__ Wt,
                                            int K, int N) {
  int idx = blockIdx.x * blockDim.x + threadIdx.x;
  if (idx >= K * N) return;
  int n = idx / K, k = idx - n * K;
  Wt[idx] = (_Float16)W[(size_t)k * N + n];
}

// --------------------- RPE MLP: one block per position row ------------------
// rows r=0..127 map to pos: 0, 1..63, 0, -63..-1
__global__ __launch_bounds__(256) void k_rpe(const float* __restrict__ pw,
                                             const float* __restrict__ pb,
                                             const float* __restrict__ lw,
                                             const float* __restrict__ lb,
                                             const float* __restrict__ ow,
                                             const float* __restrict__ ob,
                                             float* __restrict__ out) {
  __shared__ float cur[512];
  __shared__ float act[512];
  __shared__ float red[8];
  int r = blockIdx.x, tid = threadIdx.x;
  float p = (r == 0 || r == 64) ? 0.0f : (r < 64 ? (float)r : (float)(r - 128));
  for (int j = tid; j < 512; j += 256) cur[j] = relu_f(p * pw[j] + pb[j]);
  __syncthreads();

  for (int l = 0; l < 4; ++l) {
    // srms over cur (512) then relu -> act
    float s = cur[tid] * cur[tid] + cur[tid + 256] * cur[tid + 256];
    #pragma unroll
    for (int off = 16; off > 0; off >>= 1) s += __shfl_xor(s, off, 32);
    if ((tid & 31) == 0) red[tid >> 5] = s;
    __syncthreads();
    if (tid == 0) {
      float t = 0.0f;
      for (int i = 0; i < 8; ++i) t += red[i];
      red[0] = rsqrtf(t * (1.0f / 512.0f) + GTU_EPS);
    }
    __syncthreads();
    float rs = red[0];
    for (int j = tid; j < 512; j += 256) act[j] = relu_f(cur[j] * rs);
    __syncthreads();

    if (l < 3) {
      const float* w = lw + (size_t)l * 512 * 512;
      for (int j = tid; j < 512; j += 256) {
        float o = lb[l * 512 + j];
        for (int k = 0; k < 512; ++k) o += act[k] * w[(size_t)k * 512 + j];
        cur[j] = o;
      }
      __syncthreads();
    } else {
      for (int j = tid; j < 768; j += 256) {
        float o = ob[j];
        for (int k = 0; k < 512; ++k) o += act[k] * ow[(size_t)k * 768 + j];
        out[(size_t)r * 768 + j] = o;
      }
    }
  }
}

// ---------- build explicit Toeplitz matrices T[c][i][j] = a[(i-j)&127] ------
__global__ __launch_bounds__(256) void k_tbuild(const float* __restrict__ mlp,
                                                _Float16* __restrict__ T) {
  int idx = blockIdx.x * blockDim.x + threadIdx.x;  // c*4096 + i*64 + j
  int c = idx >> 12, i = (idx >> 6) & 63, j = idx & 63;
  int m = (i - j) & 127;
  T[idx] = (_Float16)mlp[(size_t)m * 768 + c];
}

// --------------- u = silu(xn @ Wu + bu) (M=32768,K=384,N=768) ---------------
__global__ __launch_bounds__(256) void k_gemm_u(const _Float16* __restrict__ xn,
                                                const _Float16* __restrict__ Wt,
                                                const float* __restrict__ bu,
                                                _Float16* __restrict__ u) {
  int lane = threadIdx.x & 31, wv = threadIdx.x >> 5;
  int i0 = blockIdx.x * 128 + wv * 16;
  int n0 = blockIdx.y * 64;
  v8f acc[4] = {vzero8(), vzero8(), vzero8(), vzero8()};
  wave_gemm16x64<384>(xn + (size_t)i0 * 384, 384, Wt + (size_t)n0 * 384, 384, lane, acc);
  int rb = i0 + ((lane >> 4) << 3);
  int c0 = n0 + (lane & 15);
  #pragma unroll
  for (int t = 0; t < 4; ++t) {
    int c = c0 + t * 16; float b = bu[c];
    #pragma unroll
    for (int r = 0; r < 8; ++r)
      u[(size_t)(rb + r) * 768 + c] = (_Float16)silu_f(acc[t][r] + b);
  }
}

// v = silu(xn @ Wv + bv), scattered to K-contiguous TNO layouts
__global__ __launch_bounds__(256) void k_gemm_v(const _Float16* __restrict__ xn,
                                                const _Float16* __restrict__ Wt,
                                                const float* __restrict__ bv,
                                                _Float16* __restrict__ vH,
                                                _Float16* __restrict__ vW) {
  int lane = threadIdx.x & 31, wv = threadIdx.x >> 5;
  int i0 = blockIdx.x * 128 + wv * 16;
  int n0 = blockIdx.y * 64;
  v8f acc[4] = {vzero8(), vzero8(), vzero8(), vzero8()};
  wave_gemm16x64<384>(xn + (size_t)i0 * 384, 384, Wt + (size_t)n0 * 384, 384, lane, acc);
  int rb = i0 + ((lane >> 4) << 3);
  int c0 = n0 + (lane & 15);
  #pragma unroll
  for (int t = 0; t < 4; ++t) {
    int c = c0 + t * 16; float b = bv[c];
    #pragma unroll
    for (int r = 0; r < 8; ++r) {
      int tok = rb + r;
      float val = silu_f(acc[t][r] + b);
      int bb = tok >> 12, hh = (tok >> 6) & 63, ww = tok & 63;
      vH[((size_t)c * 512 + bb * 64 + ww) * 64 + hh] = (_Float16)val;  // [c][b,w][h]
      vW[((size_t)c * 512 + bb * 64 + hh) * 64 + ww] = (_Float16)val;  // [c][b,h][w]
    }
  }
}

// TNO H-direction: per c, yP = TH[c] @ vH[c]  (M=64,K=64,N=512)
__global__ __launch_bounds__(256) void k_tno_h(const _Float16* __restrict__ T,
                                               const _Float16* __restrict__ vH,
                                               _Float16* __restrict__ yP) {
  int lane = threadIdx.x & 31, wv = threadIdx.x >> 5;
  int c  = blockIdx.x;
  int i0 = (wv & 3) * 16;
  int n0 = blockIdx.y * 128 + (wv >> 2) * 64;
  const _Float16* A = T  + (size_t)c * 4096;
  const _Float16* B = vH + (size_t)c * 512 * 64;
  v8f acc[4] = {vzero8(), vzero8(), vzero8(), vzero8()};
  wave_gemm16x64<64>(A + (size_t)i0 * 64, 64, B + (size_t)n0 * 64, 64, lane, acc);
  int ib = i0 + ((lane >> 4) << 3);
  int nb = n0 + (lane & 15);
  #pragma unroll
  for (int t = 0; t < 4; ++t) {
    int n = nb + t * 16; int bb = n >> 6, ww = n & 63;
    #pragma unroll
    for (int r = 0; r < 8; ++r) {
      int tok = (bb * 64 + (ib + r)) * 64 + ww;
      yP[(size_t)tok * 768 + c] = (_Float16)acc[t][r];
    }
  }
}

// TNO W-direction + sum + gate:  g = u * (yP + TW[c] @ vW[c])
__global__ __launch_bounds__(256) void k_tno_w(const _Float16* __restrict__ T,
                                               const _Float16* __restrict__ vW,
                                               const _Float16* __restrict__ yP,
                                               const _Float16* __restrict__ u,
                                               _Float16* __restrict__ g) {
  int lane = threadIdx.x & 31, wv = threadIdx.x >> 5;
  int c  = blockIdx.x;
  int i0 = (wv & 3) * 16;
  int n0 = blockIdx.y * 128 + (wv >> 2) * 64;
  const _Float16* A = T  + (size_t)c * 4096;
  const _Float16* B = vW + (size_t)c * 512 * 64;
  v8f acc[4] = {vzero8(), vzero8(), vzero8(), vzero8()};
  wave_gemm16x64<64>(A + (size_t)i0 * 64, 64, B + (size_t)n0 * 64, 64, lane, acc);
  int ib = i0 + ((lane >> 4) << 3);
  int nb = n0 + (lane & 15);
  #pragma unroll
  for (int t = 0; t < 4; ++t) {
    int n = nb + t * 16; int bb = n >> 6, hh = n & 63;
    #pragma unroll
    for (int r = 0; r < 8; ++r) {
      int tok = (bb * 64 + hh) * 64 + (ib + r);
      float val = acc[t][r] + (float)yP[(size_t)tok * 768 + c];
      g[(size_t)tok * 768 + c] = (_Float16)(val * (float)u[(size_t)tok * 768 + c]);
    }
  }
}

// final: out = g @ Wo + bo + x  (M=32768,K=768,N=384), fp32 out
__global__ __launch_bounds__(256) void k_gemm_out(const _Float16* __restrict__ g,
                                                  const _Float16* __restrict__ Wt,
                                                  const float* __restrict__ bo,
                                                  const float* __restrict__ x,
                                                  float* __restrict__ out) {
  int lane = threadIdx.x & 31, wv = threadIdx.x >> 5;
  int i0 = blockIdx.x * 128 + wv * 16;
  int n0 = blockIdx.y * 64;
  v8f acc[4] = {vzero8(), vzero8(), vzero8(), vzero8()};
  wave_gemm16x64<768>(g + (size_t)i0 * 768, 768, Wt + (size_t)n0 * 768, 768, lane, acc);
  int rb = i0 + ((lane >> 4) << 3);
  int c0 = n0 + (lane & 15);
  #pragma unroll
  for (int t = 0; t < 4; ++t) {
    int n = c0 + t * 16; float b = bo[n];
    #pragma unroll
    for (int r = 0; r < 8; ++r) {
      size_t o = (size_t)(rb + r) * 384 + n;
      out[o] = acc[t][r] + b + x[o];
    }
  }
}

// ---------------------------------------------------------------------------
extern "C" void kernel_launch(void* const* d_in, const int* in_sizes, int n_in,
                              void* d_out, int out_size, void* d_ws, size_t ws_size,
                              hipStream_t stream) {
  (void)in_sizes; (void)n_in; (void)out_size; (void)ws_size;
  const float* x   = (const float*)d_in[0];
  const float* Wu  = (const float*)d_in[3];
  const float* bu  = (const float*)d_in[4];
  const float* Wv  = (const float*)d_in[5];
  const float* bv  = (const float*)d_in[6];
  const float* Wo  = (const float*)d_in[7];
  const float* bo  = (const float*)d_in[8];
  const float* r1w[6] = {(const float*)d_in[9],  (const float*)d_in[10],
                         (const float*)d_in[11], (const float*)d_in[12],
                         (const float*)d_in[13], (const float*)d_in[14]};
  const float* r2w[6] = {(const float*)d_in[15], (const float*)d_in[16],
                         (const float*)d_in[17], (const float*)d_in[18],
                         (const float*)d_in[19], (const float*)d_in[20]};
  float* out = (float*)d_out;

  char* p = (char*)d_ws;
  auto carve = [&](size_t bytes) -> void* {
    void* r = (void*)p; p += (bytes + 255) & ~(size_t)255; return r;
  };
  _Float16* xn   = (_Float16*)carve(32768ull * 384 * 2);
  _Float16* Wut  = (_Float16*)carve(768ull * 384 * 2);
  _Float16* Wvt  = (_Float16*)carve(768ull * 384 * 2);
  _Float16* Wot  = (_Float16*)carve(384ull * 768 * 2);
  _Float16* u    = (_Float16*)carve(32768ull * 768 * 2);
  _Float16* vH   = (_Float16*)carve(768ull * 512 * 64 * 2);
  _Float16* vW   = (_Float16*)carve(768ull * 512 * 64 * 2);
  float*    mlpH = (float*)   carve(128ull * 768 * 4);
  float*    mlpW = (float*)   carve(128ull * 768 * 4);
  _Float16* TH   = (_Float16*)carve(768ull * 4096 * 2);
  _Float16* TW   = (_Float16*)carve(768ull * 4096 * 2);
  _Float16* yP   = (_Float16*)carve(32768ull * 768 * 2);
  _Float16* gg   = (_Float16*)carve(32768ull * 768 * 2);

  // weights -> f16 transposed [N][K]
  k_wt<<<dim3((384 * 768 + 255) / 256), dim3(256), 0, stream>>>(Wu, Wut, 384, 768);
  k_wt<<<dim3((384 * 768 + 255) / 256), dim3(256), 0, stream>>>(Wv, Wvt, 384, 768);
  k_wt<<<dim3((768 * 384 + 255) / 256), dim3(256), 0, stream>>>(Wo, Wot, 768, 384);

  // pre-norm
  k_srms<<<dim3(4096), dim3(256), 0, stream>>>(x, xn);

  // RPE MLPs: r2 -> H direction, r1 -> W direction
  k_rpe<<<dim3(128), dim3(256), 0, stream>>>(r2w[0], r2w[1], r2w[2], r2w[3], r2w[4], r2w[5], mlpH);
  k_rpe<<<dim3(128), dim3(256), 0, stream>>>(r1w[0], r1w[1], r1w[2], r1w[3], r1w[4], r1w[5], mlpW);
  k_tbuild<<<dim3(768 * 4096 / 256), dim3(256), 0, stream>>>(mlpH, TH);
  k_tbuild<<<dim3(768 * 4096 / 256), dim3(256), 0, stream>>>(mlpW, TW);

  // projections
  k_gemm_u<<<dim3(256, 12), dim3(256), 0, stream>>>(xn, Wut, bu, u);
  k_gemm_v<<<dim3(256, 12), dim3(256), 0, stream>>>(xn, Wvt, bv, vH, vW);

  // Toeplitz mixing (batched WMMA GEMMs), then gate
  k_tno_h<<<dim3(768, 4), dim3(256), 0, stream>>>(TH, vH, yP);
  k_tno_w<<<dim3(768, 4), dim3(256), 0, stream>>>(TW, vW, yP, u, gg);

  // output projection + bias + residual
  k_gemm_out<<<dim3(256, 6), dim3(256), 0, stream>>>(gg, Wot, bo, x, out);
}
